// SelfAttention_79680233275682
// MI455X (gfx1250) — matrix operand order repaired and checked by reading, not compiled, not run on hardware
//
#include <hip/hip_runtime.h>
#include <hip/hip_bf16.h>

typedef __bf16 bf16;
typedef __attribute__((ext_vector_type(16))) __bf16 v16bf;
typedef __attribute__((ext_vector_type(8)))  __bf16 v8bf;
typedef __attribute__((ext_vector_type(4)))  __bf16 v4bf;
typedef __attribute__((ext_vector_type(8)))  float   v8f;
typedef __attribute__((ext_vector_type(4)))  int     v4i;

#define BATCH 2
#define SEQ   2048
#define CDIM  768
#define HEADS 12
#define HD    64
#define C3    2304
#define MTOT  (BATCH*SEQ)   // 4096 tokens

// ---------------------------------------------------------------------------
// WMMA fragment layouts (ISA 7.12.2, wave32), hi = lane>>4, lr = lane&15:
//   A (16x32, bf16): lane holds row m=lr; a[j]=A[m][hi*8+j], a[8+j]=A[m][16+hi*8+j]
//   B (32x16, bf16): lane holds col n=lr; b[j]=B[hi*8+j][n], b[8+j]=B[16+hi*8+j][n]  (= row n of B^T)
//   C/D (16x16,f32): lane holds col n=lr; c[i]=C[hi*8+i][n]
// ---------------------------------------------------------------------------
__device__ __forceinline__ v8f wmma_bf16(v16bf a, v16bf b, v8f c) {
  return __builtin_amdgcn_wmma_f32_16x16x32_bf16(false, a, false, b,
                                                 (short)0, c, false, false);
}

__device__ __forceinline__ v16bf frag_a(const bf16* __restrict__ row, int hi) {
  v8bf lo = *reinterpret_cast<const v8bf*>(row + hi * 8);
  v8bf hh = *reinterpret_cast<const v8bf*>(row + 16 + hi * 8);
  v16bf a;
#pragma unroll
  for (int i = 0; i < 8; ++i) { a[i] = lo[i]; a[8 + i] = hh[i]; }
  return a;
}

__device__ __forceinline__ v16bf frag_b(const bf16* __restrict__ row, int hi) {
  v8bf lo = *reinterpret_cast<const v8bf*>(row + hi * 8);
  v8bf hh = *reinterpret_cast<const v8bf*>(row + 16 + hi * 8);
  v16bf b;
#pragma unroll
  for (int i = 0; i < 8; ++i) { b[i] = lo[i]; b[8 + i] = hh[i]; }
  return b;
}

// ---------------------------------------------------------------------------
// Async global->LDS copy (16B per lane) on the gfx1250 ASYNCcnt path.
// Builtin takes pointers to 4xint vectors (b128 payload); give it
// address-space-qualified operands (AS1 source, AS3 LDS destination).
// ---------------------------------------------------------------------------
#if __has_builtin(__builtin_amdgcn_global_load_async_to_lds_b128)
#define HAVE_ASYNC_LDS 1
typedef __attribute__((address_space(1))) v4i as1_v4i;
typedef __attribute__((address_space(3))) v4i as3_v4i;
__device__ __forceinline__ void async_cp16(const bf16* g, bf16* l) {
  __builtin_amdgcn_global_load_async_to_lds_b128(
      (as1_v4i*)g, (as3_v4i*)l, 0, 0);
}
#else
#define HAVE_ASYNC_LDS 0
__device__ __forceinline__ void async_cp16(const bf16* g, bf16* l) {
  *reinterpret_cast<v8bf*>(l) = *reinterpret_cast<const v8bf*>(g);
}
#endif

__device__ __forceinline__ void async_fence() {
#if HAVE_ASYNC_LDS
#if __has_builtin(__builtin_amdgcn_s_wait_asynccnt)
  __builtin_amdgcn_s_wait_asynccnt(0);
#else
  asm volatile("s_wait_asynccnt 0x0" ::: "memory");
#endif
#endif
}

// ---------------------------------------------------------------------------
// fp32 -> bf16 conversion, 4 elems / thread
// ---------------------------------------------------------------------------
__global__ __launch_bounds__(256)
void attnblk_cvt_bf16(const float* __restrict__ s, bf16* __restrict__ d, int n) {
  int i = (blockIdx.x * 256 + threadIdx.x) * 4;
  if (i < n) {
    float4 v = *reinterpret_cast<const float4*>(s + i);
    v4bf r;
    r[0] = (bf16)v.x; r[1] = (bf16)v.y; r[2] = (bf16)v.z; r[3] = (bf16)v.w;
    *reinterpret_cast<v4bf*>(d + i) = r;
  }
}

// ---------------------------------------------------------------------------
// QKV projection: qkv = xb @ wq^T.  32x64 register tile per wave:
// 8 WMMA per k-step against 12 b128 loads (A reused 4x, B reused 2x).
// Writes Q (pre-scaled) [B,H,N,64], K [B,H,N,64], Vt [B,H,64,N].
// ---------------------------------------------------------------------------
__global__ __launch_bounds__(256)
void attnblk_qkv_gemm(const bf16* __restrict__ xb, const bf16* __restrict__ wq,
                      bf16* __restrict__ Q, bf16* __restrict__ K,
                      bf16* __restrict__ Vt) {
  const int lane = threadIdx.x & 31, wave = threadIdx.x >> 5;
  const int hi = lane >> 4, lr = lane & 15;
  const int NT = C3 / 64;                        // 36 n-tiles
  int tile = blockIdx.x * 8 + wave;              // 128*36 = 4608 wave tiles
  int m0 = (tile / NT) * 32;
  int n0 = (tile % NT) * 64;

  const bf16* ar0 = xb + (size_t)(m0 + lr) * CDIM;
  const bf16* ar1 = xb + (size_t)(m0 + 16 + lr) * CDIM;
  const bf16* br0 = wq + (size_t)(n0 + lr) * CDIM;
  const bf16* br1 = wq + (size_t)(n0 + 16 + lr) * CDIM;
  const bf16* br2 = wq + (size_t)(n0 + 32 + lr) * CDIM;
  const bf16* br3 = wq + (size_t)(n0 + 48 + lr) * CDIM;

  v8f c[2][4] = {};
#pragma unroll 2
  for (int k = 0; k < CDIM; k += 32) {
    v16bf a0 = frag_a(ar0 + k, hi);
    v16bf a1 = frag_a(ar1 + k, hi);
    v16bf b0 = frag_b(br0 + k, hi);
    c[0][0] = wmma_bf16(a0, b0, c[0][0]);  c[1][0] = wmma_bf16(a1, b0, c[1][0]);
    v16bf b1 = frag_b(br1 + k, hi);
    c[0][1] = wmma_bf16(a0, b1, c[0][1]);  c[1][1] = wmma_bf16(a1, b1, c[1][1]);
    v16bf b2 = frag_b(br2 + k, hi);
    c[0][2] = wmma_bf16(a0, b2, c[0][2]);  c[1][2] = wmma_bf16(a1, b2, c[1][2]);
    v16bf b3 = frag_b(br3 + k, hi);
    c[0][3] = wmma_bf16(a0, b3, c[0][3]);  c[1][3] = wmma_bf16(a1, b3, c[1][3]);
  }

  const int h = n0 / 192, sq = (n0 / 64) % 3;
  if (sq == 0) {                                 // Q, fold softmax scale
#pragma unroll
    for (int mi = 0; mi < 2; ++mi)
#pragma unroll
      for (int s = 0; s < 4; ++s)
#pragma unroll
        for (int i = 0; i < 8; ++i) {
          int tg = m0 + mi * 16 + hi * 8 + i, b = tg >> 11, ns = tg & (SEQ - 1);
          Q[(((size_t)(b * HEADS + h)) * SEQ + ns) * HD + s * 16 + lr] =
              (bf16)(c[mi][s][i] * 0.125f);
        }
  } else if (sq == 1) {                          // K
#pragma unroll
    for (int mi = 0; mi < 2; ++mi)
#pragma unroll
      for (int s = 0; s < 4; ++s)
#pragma unroll
        for (int i = 0; i < 8; ++i) {
          int tg = m0 + mi * 16 + hi * 8 + i, b = tg >> 11, ns = tg & (SEQ - 1);
          K[(((size_t)(b * HEADS + h)) * SEQ + ns) * HD + s * 16 + lr] =
              (bf16)c[mi][s][i];
        }
  } else {                                       // V -> transposed [hd][N]
#pragma unroll
    for (int mi = 0; mi < 2; ++mi) {
      int tg0 = m0 + mi * 16 + hi * 8, b = tg0 >> 11, ns0 = tg0 & (SEQ - 1);
#pragma unroll
      for (int s = 0; s < 4; ++s) {
        v8bf pk;
#pragma unroll
        for (int i = 0; i < 8; ++i) pk[i] = (bf16)c[mi][s][i];
        *reinterpret_cast<v8bf*>(
            Vt + (((size_t)(b * HEADS + h)) * HD + s * 16 + lr) * SEQ + ns0) = pk;
      }
    }
  }
}

// ---------------------------------------------------------------------------
// Flash attention. One block = 8 waves sharing one (b,h); each wave owns a
// 16-query tile. KV streamed in 32-wide chunks, double-buffered in LDS via
// async global->LDS copies (copy of chunk c+1 overlaps compute of chunk c).
// P (16x32) is transposed C-layout -> A-layout through per-wave LDS.
// ---------------------------------------------------------------------------
__global__ __launch_bounds__(256)
void attnblk_flash(const bf16* __restrict__ Q, const bf16* __restrict__ Kb,
                   const bf16* __restrict__ Vt, bf16* __restrict__ AO) {
  __shared__ __align__(16) bf16 Kl[2][32][64];   // 8 KB: kv x d
  __shared__ __align__(16) bf16 Vl[2][64][32];   // 8 KB: d  x kv
  __shared__ __align__(16) bf16 Pl[8][16][32];   // 8 KB: per-wave P tiles

  const int tid  = threadIdx.x;
  const int lane = tid & 31, wave = tid >> 5;
  const int hi = lane >> 4, lr = lane & 15;

  size_t bh = (size_t)(blockIdx.x >> 4);               // 24 (b,h) pairs
  int m0 = ((blockIdx.x & 15) * 8 + wave) * 16;        // 128 m-tiles per (b,h)

  const bf16* qrow  = Q  + (bh * SEQ + m0 + lr) * HD;
  const bf16* kbase = Kb + bh * SEQ * HD;
  const bf16* vbase = Vt + bh * HD * SEQ;

  // cooperative copy mapping: 16B per thread per matrix
  const int kr = tid >> 3, kseg = (tid & 7) * 8;       // K chunk 32x64
  const int vr = tid >> 2, vseg = (tid & 3) * 8;       // V chunk 64x32

  async_cp16(kbase + (size_t)kr * HD + kseg, &Kl[0][kr][kseg]);
  async_cp16(vbase + (size_t)vr * SEQ + vseg, &Vl[0][vr][vseg]);

  v16bf aq0 = frag_a(qrow, hi);                        // d = 0..31
  v16bf aq1 = frag_a(qrow + 32, hi);                   // d = 32..63

  float mrow[8], lrow[8];
  v8f acc[4] = {v8f{}, v8f{}, v8f{}, v8f{}};
#pragma unroll
  for (int i = 0; i < 8; ++i) { mrow[i] = -1e30f; lrow[i] = 0.f; }

  async_fence();
  __syncthreads();

  const int NCHUNK = SEQ / 32;                         // 64
  for (int ch = 0; ch < NCHUNK; ++ch) {
    int p = ch & 1;
    if (ch + 1 < NCHUNK) {                             // prefetch next chunk
      int kv = (ch + 1) * 32;
      async_cp16(kbase + (size_t)(kv + kr) * HD + kseg, &Kl[p ^ 1][kr][kseg]);
      async_cp16(vbase + (size_t)vr * SEQ + kv + vseg, &Vl[p ^ 1][vr][vseg]);
    }

    v8f s0 = {}, s1 = {};
    s0 = wmma_bf16(aq0, frag_b(&Kl[p][lr][0], hi), s0);
    s0 = wmma_bf16(aq1, frag_b(&Kl[p][lr][32], hi), s0);
    s1 = wmma_bf16(aq0, frag_b(&Kl[p][16 + lr][0], hi), s1);
    s1 = wmma_bf16(aq1, frag_b(&Kl[p][16 + lr][32], hi), s1);

    // online softmax; row r=hi*8+i lives across the 16 lanes of each half
#pragma unroll
    for (int i = 0; i < 8; ++i) {
      float cm = fmaxf(s0[i], s1[i]);
      cm = fmaxf(cm, __shfl_xor(cm, 1, 32));
      cm = fmaxf(cm, __shfl_xor(cm, 2, 32));
      cm = fmaxf(cm, __shfl_xor(cm, 4, 32));
      cm = fmaxf(cm, __shfl_xor(cm, 8, 32));
      float mn = fmaxf(mrow[i], cm);
      float alpha = __expf(mrow[i] - mn);
      float p0 = __expf(s0[i] - mn);
      float p1 = __expf(s1[i] - mn);
      float rs = p0 + p1;
      rs += __shfl_xor(rs, 1, 32);
      rs += __shfl_xor(rs, 2, 32);
      rs += __shfl_xor(rs, 4, 32);
      rs += __shfl_xor(rs, 8, 32);
      lrow[i] = lrow[i] * alpha + rs;
      mrow[i] = mn;
#pragma unroll
      for (int t = 0; t < 4; ++t) acc[t][i] *= alpha;
      Pl[wave][hi * 8 + i][lr]      = (bf16)p0;
      Pl[wave][hi * 8 + i][16 + lr] = (bf16)p1;
    }
    asm volatile("s_wait_dscnt 0x0" ::: "memory");     // P visible to frag load

    v16bf ap = frag_a(&Pl[wave][lr][0], hi);
#pragma unroll
    for (int t = 0; t < 4; ++t)
      acc[t] = wmma_bf16(ap, frag_b(&Vl[p][t * 16 + lr][0], hi), acc[t]);

    async_fence();                                     // next buffers landed
    __syncthreads();                                   // all waves done with p
  }

#pragma unroll
  for (int i = 0; i < 8; ++i) {
    float inv = 1.0f / lrow[i];
    size_t row = (size_t)((bh / HEADS) * SEQ + m0 + hi * 8 + i) * CDIM +
                 (bh % HEADS) * HD + lr;
#pragma unroll
    for (int t = 0; t < 4; ++t)
      AO[row + t * 16] = (bf16)(acc[t][i] * inv);
  }
}

// ---------------------------------------------------------------------------
// Output projection: out = AO @ wm^T + bias (fp32 out). 32x64 wave tile.
// ---------------------------------------------------------------------------
__global__ __launch_bounds__(256)
void attnblk_mlp_gemm(const bf16* __restrict__ AO, const bf16* __restrict__ wm,
                      const float* __restrict__ bias, float* __restrict__ out) {
  const int lane = threadIdx.x & 31, wave = threadIdx.x >> 5;
  const int hi = lane >> 4, lr = lane & 15;
  const int NT = CDIM / 64;                      // 12 n-tiles
  int tile = blockIdx.x * 8 + wave;              // 128*12 = 1536 wave tiles
  int m0 = (tile / NT) * 32;
  int n0 = (tile % NT) * 64;

  const bf16* ar0 = AO + (size_t)(m0 + lr) * CDIM;
  const bf16* ar1 = AO + (size_t)(m0 + 16 + lr) * CDIM;
  const bf16* br0 = wm + (size_t)(n0 + lr) * CDIM;
  const bf16* br1 = wm + (size_t)(n0 + 16 + lr) * CDIM;
  const bf16* br2 = wm + (size_t)(n0 + 32 + lr) * CDIM;
  const bf16* br3 = wm + (size_t)(n0 + 48 + lr) * CDIM;

  v8f c[2][4] = {};
#pragma unroll 2
  for (int k = 0; k < CDIM; k += 32) {
    v16bf a0 = frag_a(ar0 + k, hi);
    v16bf a1 = frag_a(ar1 + k, hi);
    v16bf b0 = frag_b(br0 + k, hi);
    c[0][0] = wmma_bf16(a0, b0, c[0][0]);  c[1][0] = wmma_bf16(a1, b0, c[1][0]);
    v16bf b1 = frag_b(br1 + k, hi);
    c[0][1] = wmma_bf16(a0, b1, c[0][1]);  c[1][1] = wmma_bf16(a1, b1, c[1][1]);
    v16bf b2 = frag_b(br2 + k, hi);
    c[0][2] = wmma_bf16(a0, b2, c[0][2]);  c[1][2] = wmma_bf16(a1, b2, c[1][2]);
    v16bf b3 = frag_b(br3 + k, hi);
    c[0][3] = wmma_bf16(a0, b3, c[0][3]);  c[1][3] = wmma_bf16(a1, b3, c[1][3]);
  }

#pragma unroll
  for (int s = 0; s < 4; ++s) {
    float bv = bias[n0 + s * 16 + lr];
#pragma unroll
    for (int mi = 0; mi < 2; ++mi)
#pragma unroll
      for (int i = 0; i < 8; ++i)
        out[(size_t)(m0 + mi * 16 + hi * 8 + i) * CDIM + n0 + s * 16 + lr] =
            c[mi][s][i] + bv;
  }
}

// ---------------------------------------------------------------------------
extern "C" void kernel_launch(void* const* d_in, const int* in_sizes, int n_in,
                              void* d_out, int out_size, void* d_ws, size_t ws_size,
                              hipStream_t stream) {
  const float* x     = (const float*)d_in[0];
  const float* w_qkv = (const float*)d_in[1];
  const float* w_mlp = (const float*)d_in[2];
  const float* b_mlp = (const float*)d_in[3];
  float* out = (float*)d_out;

  char* ws = (char*)d_ws;
  size_t o = 0;
  bf16* xb  = (bf16*)(ws + o); o += (size_t)MTOT * CDIM * 2;
  bf16* wqb = (bf16*)(ws + o); o += (size_t)C3 * CDIM * 2;
  bf16* wmb = (bf16*)(ws + o); o += (size_t)CDIM * CDIM * 2;
  bf16* Qb  = (bf16*)(ws + o); o += (size_t)BATCH * HEADS * SEQ * HD * 2;
  bf16* Kb  = (bf16*)(ws + o); o += (size_t)BATCH * HEADS * SEQ * HD * 2;
  bf16* Vtb = (bf16*)(ws + o); o += (size_t)BATCH * HEADS * SEQ * HD * 2;
  bf16* AOb = (bf16*)(ws + o); o += (size_t)MTOT * CDIM * 2;   // ~34.5 MB total

  int nx = MTOT * CDIM, nq = C3 * CDIM, nm = CDIM * CDIM;
  attnblk_cvt_bf16<<<(nx / 4 + 255) / 256, 256, 0, stream>>>(x, xb, nx);
  attnblk_cvt_bf16<<<(nq / 4 + 255) / 256, 256, 0, stream>>>(w_qkv, wqb, nq);
  attnblk_cvt_bf16<<<(nm / 4 + 255) / 256, 256, 0, stream>>>(w_mlp, wmb, nm);

  attnblk_qkv_gemm<<<576, 256, 0, stream>>>(xb, wqb, Qb, Kb, Vtb);   // 4608/8
  attnblk_flash<<<384, 256, 0, stream>>>(Qb, Kb, Vtb, AOb);          // 3072/8
  attnblk_mlp_gemm<<<192, 256, 0, stream>>>(AOb, wmb, b_mlp, out);   // 1536/8
}
